// STVQVae_10307921510462
// MI455X (gfx1250) — compile-verified, matrix operand order checked by reading, not run on hardware
//
#include <hip/hip_runtime.h>

// ---- model dims ----
#define BB   2
#define TT   16
#define NPP  256
#define DD   768
#define NHH  12
#define DHH  64
#define DLL  3072
#define CSS  1024
#define LDD  256
#define MM   8192      // B*T*NP tokens

typedef __bf16 bf16;
typedef __bf16 v16bf __attribute__((ext_vector_type(16)));
typedef __bf16 v8bf  __attribute__((ext_vector_type(8)));
typedef float  v8f   __attribute__((ext_vector_type(8)));

static inline int cdiv(int a, int b) { return (a + b - 1) / b; }

__device__ __forceinline__ float gelu_f(float x) {
    float x3 = x * x * x;
    return 0.5f * x * (1.f + tanhf(0.79788456080286536f * (x + 0.044715f * x3)));
}

// ======================= WMMA GEMM =======================
// C[M,N] = alpha * A[M,K] @ B^T  (+ bias, + resid, + gelu)
// A bf16 row-major [M,K]; B bf16 row-major [N,K]  (all operands pre-arranged N-major).
// 256 threads = 8 waves (4x2), block tile 128x128, wave tile 32x64, K-step 32.
// All K are multiples of 8 -> bounds handled per 8-element segment (full in or full out).
template <bool GELU, bool RES, bool WF32, bool WB16>
__global__ __launch_bounds__(256) void k_gemm(
    const bf16* __restrict__ A, const bf16* __restrict__ Bm,
    const float* __restrict__ bias, const float* __restrict__ resid,
    float* __restrict__ Cf, bf16* __restrict__ Cb,
    int M, int N, int K, float alpha, long sA, long sB, long sC)
{
    __shared__ __align__(16) bf16 As[128][40];  // [m][k], 80B rows (16B multiple)
    __shared__ __align__(16) bf16 Bs[128][40];  // [n][k]

    const int tid = threadIdx.x;
    const long bz = blockIdx.z;
    A  += bz * sA;
    Bm += bz * sB;
    const long cOff = bz * sC;

    const int m0 = blockIdx.y * 128;
    const int n0 = blockIdx.x * 128;
    const int wid  = tid >> 5;
    const int lane = tid & 31;
    const int wm = wid >> 1, wn = wid & 1;        // 4 x 2 wave grid
    const int half = lane >> 4, lr = lane & 15;

    v8f acc[2][4];
#pragma unroll
    for (int i = 0; i < 2; ++i)
#pragma unroll
        for (int j = 0; j < 4; ++j)
#pragma unroll
            for (int e = 0; e < 8; ++e) acc[i][j][e] = 0.f;

    v8bf zer;
#pragma unroll
    for (int i = 0; i < 8; ++i) zer[i] = (bf16)0.f;

    // per-thread tile-load mapping: 512 segments of 8, 2 per thread
    // idx -> row = idx>>2 (0..127), ks = (idx&3)*8 (0,8,16,24)
    for (int k0 = 0; k0 < K; k0 += 32) {
#pragma unroll
        for (int ss = 0; ss < 2; ++ss) {
            int idx = tid + ss * 256;
            int row = idx >> 2;
            int ks  = (idx & 3) * 8;
            int gk  = k0 + ks;
            // A segment
            {
                int gr = m0 + row;
                const bf16* src = A + (long)gr * K + gk;
                v8bf v = zer;
                if (gr < M && gk + 8 <= K) {
                    v = *(const v8bf*)src;
                    if (k0 + 32 < K) __builtin_prefetch(src + 32, 0, 1);
                }
                *(v8bf*)&As[row][ks] = v;
            }
            // B segment (B is [N,K])
            {
                int gn = n0 + row;
                const bf16* src = Bm + (long)gn * K + gk;
                v8bf v = zer;
                if (gn < N && gk + 8 <= K) {
                    v = *(const v8bf*)src;
                    if (k0 + 32 < K) __builtin_prefetch(src + 32, 0, 1);
                }
                *(v8bf*)&Bs[row][ks] = v;
            }
        }
        __syncthreads();

        // fragments (ISA 16-bit layout: lanes<16 hold K0-7/16-23, lanes>=16 hold K8-15/24-31)
        v16bf af[2], bfr[4];
#pragma unroll
        for (int s = 0; s < 2; ++s) {
            int r = wm * 32 + s * 16 + lr;
            v8bf lo = *(const v8bf*)&As[r][8 * half];
            v8bf hi = *(const v8bf*)&As[r][16 + 8 * half];
#pragma unroll
            for (int i = 0; i < 8; ++i) { af[s][i] = lo[i]; af[s][i + 8] = hi[i]; }
        }
#pragma unroll
        for (int s = 0; s < 4; ++s) {
            int c = wn * 64 + s * 16 + lr;
            v8bf lo = *(const v8bf*)&Bs[c][8 * half];
            v8bf hi = *(const v8bf*)&Bs[c][16 + 8 * half];
#pragma unroll
            for (int i = 0; i < 8; ++i) { bfr[s][i] = lo[i]; bfr[s][i + 8] = hi[i]; }
        }
#pragma unroll
        for (int ms = 0; ms < 2; ++ms)
#pragma unroll
            for (int ns = 0; ns < 4; ++ns)
                acc[ms][ns] = __builtin_amdgcn_wmma_f32_16x16x32_bf16(
                    false, af[ms], false, bfr[ns], (short)0, acc[ms][ns], false, false);
        __syncthreads();
    }

    // ---- epilogue ----
#pragma unroll
    for (int ms = 0; ms < 2; ++ms)
#pragma unroll
        for (int ns = 0; ns < 4; ++ns) {
            int gcol = n0 + wn * 64 + ns * 16 + lr;
            if (gcol >= N) continue;
            float bv = (bias != nullptr) ? bias[gcol] : 0.f;
#pragma unroll
            for (int v = 0; v < 8; ++v) {
                int grow = m0 + wm * 32 + ms * 16 + half * 8 + v;
                if (grow >= M) continue;
                long off = cOff + (long)grow * N + gcol;
                float val = alpha * acc[ms][ns][v] + bv;
                if (RES)  val += resid[off];
                if (GELU) val = gelu_f(val);
                if (WF32) Cf[off] = val;
                if (WB16) Cb[off] = (bf16)val;
            }
        }
}

// ======================= elementwise kernels =======================
__global__ void k_cvt(const float* __restrict__ in, bf16* __restrict__ out, long n) {
    long i = (long)blockIdx.x * blockDim.x + threadIdx.x;
    long st = (long)gridDim.x * blockDim.x;
    for (; i < n; i += st) out[i] = (bf16)in[i];
}

// f32 [K,N] -> bf16 [N,K]  (LDS 32x32 tile transpose, both sides coalesced)
__global__ __launch_bounds__(256) void k_cvt_t(const float* __restrict__ in,
                                               bf16* __restrict__ out, int K, int N) {
    __shared__ float tile[32][33];
    int tx = threadIdx.x & 31;
    int ty = threadIdx.x >> 5;     // 0..7
    int k0 = blockIdx.y * 32;
    int n0 = blockIdx.x * 32;
#pragma unroll
    for (int r = ty; r < 32; r += 8) {
        int gk = k0 + r, gn = n0 + tx;
        tile[r][tx] = (gk < K && gn < N) ? in[(long)gk * N + gn] : 0.f;
    }
    __syncthreads();
#pragma unroll
    for (int r = ty; r < 32; r += 8) {
        int gn = n0 + r, gk = k0 + tx;
        if (gn < N && gk < K) out[(long)gn * K + gk] = (bf16)tile[tx][r];
    }
}

__global__ void k_patchify(const float* __restrict__ x, const float* __restrict__ ps,
                           const float* __restrict__ pt, float* __restrict__ tok) {
    long n = (long)MM * DD;
    long i = (long)blockIdx.x * blockDim.x + threadIdx.x;
    long st = (long)gridDim.x * blockDim.x;
    for (; i < n; i += st) {
        int m = (int)(i / DD), f = (int)(i % DD);
        int b = m / (TT * NPP);
        int t = (m / NPP) % TT;
        int np = m % NPP;
        int c = f >> 8;
        int py = (f >> 4) & 15;
        int px = f & 15;
        int ph = np >> 4, pw = np & 15;
        long src = ((long)(b * 3 + c) * TT + t) * 65536 + (ph * 16 + py) * 256 + (pw * 16 + px);
        tok[i] = x[src] + ps[np * DD + f] + pt[t * DD + f];
    }
}

__global__ void k_unpatch(const float* __restrict__ h, float* __restrict__ out) {
    long n = (long)MM * DD;
    long i = (long)blockIdx.x * blockDim.x + threadIdx.x;
    long st = (long)gridDim.x * blockDim.x;
    for (; i < n; i += st) {
        int m = (int)(i / DD), f = (int)(i % DD);
        int b = m / (TT * NPP);
        int t = (m / NPP) % TT;
        int np = m % NPP;
        int c = f >> 8;
        int py = (f >> 4) & 15;
        int px = f & 15;
        int ph = np >> 4, pw = np & 15;
        long dst = ((long)(b * 3 + c) * TT + t) * 65536 + (ph * 16 + py) * 256 + (pw * 16 + px);
        out[dst] = h[i];
    }
}

__global__ __launch_bounds__(256) void k_ln(const float* __restrict__ x,
                                            const float* __restrict__ g,
                                            const float* __restrict__ b,
                                            bf16* __restrict__ y, int cols) {
    __shared__ float red[256];
    int row = blockIdx.x, tid = threadIdx.x;
    const float* xr = x + (long)row * cols;
    float s = 0.f, s2 = 0.f;
    for (int c = tid; c < cols; c += 256) { float v = xr[c]; s += v; s2 += v * v; }
    red[tid] = s; __syncthreads();
    for (int o = 128; o > 0; o >>= 1) { if (tid < o) red[tid] += red[tid + o]; __syncthreads(); }
    float mean = red[0] / cols; __syncthreads();
    red[tid] = s2; __syncthreads();
    for (int o = 128; o > 0; o >>= 1) { if (tid < o) red[tid] += red[tid + o]; __syncthreads(); }
    float var = red[0] / cols - mean * mean;
    float inv = rsqrtf(var + 1e-5f);
    bf16* yr = y + (long)row * cols;
    for (int c = tid; c < cols; c += 256)
        yr[c] = (bf16)((xr[c] - mean) * inv * g[c] + b[c]);
}

__global__ __launch_bounds__(256) void k_softmax(const float* __restrict__ S,
                                                 bf16* __restrict__ P, int seq, int causal) {
    __shared__ float red[256];
    int row = blockIdx.x, tid = threadIdx.x;
    int q = row % seq;
    int lim = causal ? (q + 1) : seq;
    const float* sr = S + (long)row * seq;
    bf16* pr = P + (long)row * seq;
    float mx = -3.4e38f;
    for (int c = tid; c < lim; c += 256) mx = fmaxf(mx, sr[c]);
    red[tid] = mx; __syncthreads();
    for (int o = 128; o > 0; o >>= 1) { if (tid < o) red[tid] = fmaxf(red[tid], red[tid + o]); __syncthreads(); }
    mx = red[0]; __syncthreads();
    float sum = 0.f;
    for (int c = tid; c < lim; c += 256) sum += expf(sr[c] - mx);
    red[tid] = sum; __syncthreads();
    for (int o = 128; o > 0; o >>= 1) { if (tid < o) red[tid] += red[tid + o]; __syncthreads(); }
    float inv = 1.f / red[0];
    for (int c = tid; c < seq; c += 256)
        pr[c] = (bf16)((c < lim) ? expf(sr[c] - mx) * inv : 0.f);
}

// pack [M,768] f32 -> bf16 heads; spatial: BH=b*t*h, S=NP; temporal: BH=b*n*h, S=T
// trans=0: [BH, S, 64] ; trans=1: [BH, 64, S] (for the V operand: B=[N,K])
__global__ void k_pack(const float* __restrict__ src, bf16* __restrict__ dst,
                       int spatial, int trans) {
    long n = (long)MM * DD;
    long i = (long)blockIdx.x * blockDim.x + threadIdx.x;
    long st = (long)gridDim.x * blockDim.x;
    int S = spatial ? NPP : TT;
    for (; i < n; i += st) {
        int m = (int)(i / DD), f = (int)(i % DD);
        int h = f / DHH, dd = f % DHH;
        int b = m / (TT * NPP), t = (m / NPP) % TT, np = m % NPP;
        long bh = spatial ? ((long)(b * TT + t) * NHH + h)
                          : ((long)(b * NPP + np) * NHH + h);
        int sq = spatial ? np : t;
        long d = trans ? ((bh * DHH + dd) * S + sq)
                       : ((bh * S + sq) * DHH + dd);
        dst[d] = (bf16)src[i];
    }
}

__global__ void k_unpack(const float* __restrict__ src, bf16* __restrict__ dst, int spatial) {
    long n = (long)MM * DD;
    long i = (long)blockIdx.x * blockDim.x + threadIdx.x;
    long st = (long)gridDim.x * blockDim.x;
    int S = spatial ? NPP : TT;
    for (; i < n; i += st) {
        int m = (int)(i / DD), f = (int)(i % DD);
        int h = f / DHH, dd = f % DHH;
        int b = m / (TT * NPP), t = (m / NPP) % TT, np = m % NPP;
        long bh = spatial ? ((long)(b * TT + t) * NHH + h)
                          : ((long)(b * NPP + np) * NHH + h);
        int sq = spatial ? np : t;
        dst[i] = (bf16)src[(bh * S + sq) * DHH + dd];
    }
}

__global__ void k_cnorm(const float* __restrict__ cb, float* __restrict__ cn) {
    int j = blockIdx.x * blockDim.x + threadIdx.x;
    if (j < CSS) {
        const float* r = cb + (long)j * LDD;
        float s = 0.f;
        for (int c = 0; c < LDD; ++c) s += r[c] * r[c];
        cn[j] = s;
    }
}

// argmin_j (||c_j||^2 - 2 z.c_j) ; gather codebook row -> z_q (f32 + bf16)
__global__ __launch_bounds__(256) void k_vq(const float* __restrict__ Sc,
                                            const float* __restrict__ cn,
                                            const float* __restrict__ cb,
                                            float* __restrict__ zq, bf16* __restrict__ zqb) {
    __shared__ float smin[256];
    __shared__ int   sidx[256];
    int row = blockIdx.x, tid = threadIdx.x;
    const float* sr = Sc + (long)row * CSS;
    float bm = 3.4e38f; int bj = 0;
    for (int j = tid; j < CSS; j += 256) {
        float d = cn[j] - 2.f * sr[j];
        if (d < bm) { bm = d; bj = j; }
    }
    smin[tid] = bm; sidx[tid] = bj; __syncthreads();
    for (int o = 128; o > 0; o >>= 1) {
        if (tid < o) {
            float om = smin[tid + o]; int oj = sidx[tid + o];
            if (om < smin[tid] || (om == smin[tid] && oj < sidx[tid])) { smin[tid] = om; sidx[tid] = oj; }
        }
        __syncthreads();
    }
    int best = sidx[0];
    const float* cr = cb + (long)best * LDD;
    for (int c = tid; c < LDD; c += 256) {
        float v = cr[c];
        zq[(long)row * LDD + c]  = v;
        zqb[(long)row * LDD + c] = (bf16)v;
    }
}

// ======================= host-side orchestration =======================
struct Ctx {
    const float* const* in;
    float *X, *AO, *ATT, *cn;
    bf16 *Pb, *LNB, *Qb, *Kb, *Vb, *Hb, *WB, *ZEb, *ZQb;
    hipStream_t st;
};

static void gemmNN(const bf16* A, const bf16* Bm, const float* bias, float* Cf,
                   int M, int N, int K, int batch, long sA, long sB, long sC,
                   float alpha, hipStream_t st) {
    dim3 g(cdiv(N, 128), cdiv(M, 128), batch);
    k_gemm<false, false, true, false><<<g, 256, 0, st>>>(A, Bm, bias, nullptr, Cf, nullptr, M, N, K, alpha, sA, sB, sC);
}
static void gemmNN_res(const bf16* A, const bf16* Bm, const float* bias, const float* resid,
                       float* Cf, int M, int N, int K, hipStream_t st) {
    dim3 g(cdiv(N, 128), cdiv(M, 128), 1);
    k_gemm<false, true, true, false><<<g, 256, 0, st>>>(A, Bm, bias, resid, Cf, nullptr, M, N, K, 1.f, 0, 0, 0);
}
static void gemmNN_gelu(const bf16* A, const bf16* Bm, const float* bias, bf16* Cb,
                        int M, int N, int K, hipStream_t st) {
    dim3 g(cdiv(N, 128), cdiv(M, 128), 1);
    k_gemm<true, false, false, true><<<g, 256, 0, st>>>(A, Bm, bias, nullptr, nullptr, Cb, M, N, K, 1.f, 0, 0, 0);
}
static void gemmNN_dual(const bf16* A, const bf16* Bm, const float* bias, float* Cf, bf16* Cb,
                        int M, int N, int K, hipStream_t st) {
    dim3 g(cdiv(N, 128), cdiv(M, 128), 1);
    k_gemm<false, false, true, true><<<g, 256, 0, st>>>(A, Bm, bias, nullptr, Cf, Cb, M, N, K, 1.f, 0, 0, 0);
}
// transpose-convert weight [K,N] f32 -> [N,K] bf16
static void cvtT(const float* w, bf16* dst, int K, int N, hipStream_t st) {
    dim3 g(cdiv(N, 32), cdiv(K, 32), 1);
    k_cvt_t<<<g, 256, 0, st>>>(w, dst, K, N);
}

// wqi: input index of q.w ; spatial in {0,1}; causal in {0,1}
static void run_attn(Ctx& p, int wqi, int spatial, int causal) {
    const int BH = spatial ? (BB * TT * NHH) : (BB * NPP * NHH);  // 384 / 6144
    const int S  = spatial ? NPP : TT;                             // 256 / 16
    const int CH = spatial ? 96 : 1536;                            // batch chunk
    bf16* qkv[3] = { p.Qb, p.Kb, p.Vb };
    for (int i = 0; i < 3; ++i) {
        cvtT(p.in[wqi + 2 * i], p.WB, DD, DD, p.st);
        gemmNN(p.LNB, p.WB, p.in[wqi + 2 * i + 1], p.AO, MM, DD, DD, 1, 0, 0, 0, 1.f, p.st);
        k_pack<<<2048, 256, 0, p.st>>>(p.AO, qkv[i], spatial, (i == 2) ? 1 : 0);
    }
    const float scale = 0.125f;  // dh^-0.5, dh = 64
    for (int ch = 0; ch * CH < BH; ++ch) {
        long off = (long)ch * CH * S * DHH;
        // scores: Q[s,dh] @ K[s,dh]^T   (B = K, already [N=s, K=dh])
        gemmNN(p.Qb + off, p.Kb + off, nullptr, p.ATT, S, S, DHH, CH,
               (long)S * DHH, (long)S * DHH, (long)S * S, scale, p.st);
        k_softmax<<<CH * S, 256, 0, p.st>>>(p.ATT, p.Pb, S, causal);
        // out: P[s,s] @ V[s,dh]         (B = V packed transposed [N=dh, K=s])
        gemmNN(p.Pb, p.Vb + off, nullptr, p.AO + off, S, DHH, S, CH,
               (long)S * S, (long)S * DHH, (long)S * DHH, 1.f, p.st);
    }
    k_unpack<<<2048, 256, 0, p.st>>>(p.AO, p.LNB, spatial);
    cvtT(p.in[wqi + 6], p.WB, DD, DD, p.st);
    gemmNN_res(p.LNB, p.WB, p.in[wqi + 7], p.X, p.X, MM, DD, DD, p.st);
}

// layer params (26 tensors): 0 ln_s.g 1 ln_s.b 2..9 attn_s 10 ln_t.g 11 ln_t.b
// 12..19 attn_t 20 ln_m.g 21 ln_m.b 22 mlp1.w 23 mlp1.b 24 mlp2.w 25 mlp2.b
static void st_layer(Ctx& p, int base, int causal) {
    k_ln<<<MM, 256, 0, p.st>>>(p.X, p.in[base + 0], p.in[base + 1], p.LNB, DD);
    run_attn(p, base + 2, 1, 0);
    k_ln<<<MM, 256, 0, p.st>>>(p.X, p.in[base + 10], p.in[base + 11], p.LNB, DD);
    run_attn(p, base + 12, 0, causal);
    k_ln<<<MM, 256, 0, p.st>>>(p.X, p.in[base + 20], p.in[base + 21], p.LNB, DD);
    cvtT(p.in[base + 22], p.WB, DD, DLL, p.st);
    gemmNN_gelu(p.LNB, p.WB, p.in[base + 23], p.Hb, MM, DLL, DD, p.st);
    cvtT(p.in[base + 24], p.WB, DLL, DD, p.st);
    gemmNN_res(p.Hb, p.WB, p.in[base + 25], p.X, p.X, MM, DD, DLL, p.st);
}

extern "C" void kernel_launch(void* const* d_in, const int* in_sizes, int n_in,
                              void* d_out, int out_size, void* d_ws, size_t ws_size,
                              hipStream_t stream) {
    (void)in_sizes; (void)n_in; (void)out_size; (void)ws_size;
    const float* const* in = (const float* const*)d_in;

    char* w = (char*)d_ws;
    auto take = [&](size_t bytes) { char* r = w; w += (bytes + 255) & ~(size_t)255; return r; };

    Ctx p{};
    p.in = in; p.st = stream;
    p.X   = (float*)take((size_t)MM * DD * 4);            // master activations
    p.AO  = (float*)take((size_t)MM * DD * 4);            // qkv f32 / attn out / dec_out
    p.ATT = (float*)take((size_t)MM * CSS * 4);           // score chunks / VQ distances
    p.cn  = (float*)take((size_t)CSS * 4);
    p.Pb  = (bf16*)take((size_t)96 * 256 * 256 * 2);      // softmax probs (chunked)
    p.LNB = (bf16*)take((size_t)MM * DD * 2);             // LN output / packed attn-out
    p.Qb  = (bf16*)take((size_t)MM * DD * 2);
    p.Kb  = (bf16*)take((size_t)MM * DD * 2);
    p.Vb  = (bf16*)take((size_t)MM * DD * 2);
    p.Hb  = (bf16*)take((size_t)MM * DLL * 2);            // MLP hidden (bf16 only)
    p.WB  = (bf16*)take((size_t)DLL * DD * 2);            // transposed-weight slot
    p.ZEb = (bf16*)take((size_t)MM * LDD * 2);
    p.ZQb = (bf16*)take((size_t)MM * LDD * 2);

    float* out      = (float*)d_out;
    float* outRecon = out;                                  // 6,291,456 = MM*DD
    float* outZe    = out + (size_t)MM * DD;
    float* outZq    = outZe + (size_t)MM * LDD;

    // ---- encoder ----
    k_patchify<<<2048, 256, 0, stream>>>(in[0], in[1], in[2], p.X);
    st_layer(p, 3, 1);    // enc layer 0 (temporal causal)
    st_layer(p, 29, 1);   // enc layer 1
    k_ln<<<MM, 256, 0, stream>>>(p.X, in[55], in[56], p.LNB, DD);
    cvtT(in[57], p.WB, DD, LDD, stream);
    gemmNN_dual(p.LNB, p.WB, in[58], outZe, p.ZEb, MM, LDD, DD, stream);  // z_e

    // ---- vector quantization ----
    k_cnorm<<<cdiv(CSS, 256), 256, 0, stream>>>(in[59], p.cn);
    k_cvt<<<1024, 256, 0, stream>>>(in[59], p.WB, (long)CSS * LDD);       // codebook already [N,K]
    gemmNN(p.ZEb, p.WB, nullptr, p.ATT, MM, CSS, LDD, 1, 0, 0, 0, 1.f, stream);  // z.c^T
    k_vq<<<MM, 256, 0, stream>>>(p.ATT, p.cn, in[59], outZq, p.ZQb);

    // ---- decoder ----
    cvtT(in[60], p.WB, LDD, DD, stream);
    gemmNN(p.ZQb, p.WB, in[61], p.X, MM, DD, LDD, 1, 0, 0, 0, 1.f, stream);
    st_layer(p, 62, 0);   // dec layer 0 (non-causal)
    st_layer(p, 88, 0);   // dec layer 1
    k_ln<<<MM, 256, 0, stream>>>(p.X, in[114], in[115], p.LNB, DD);
    cvtT(in[116], p.WB, DD, DD, stream);
    gemmNN(p.LNB, p.WB, in[117], p.AO, MM, DD, DD, 1, 0, 0, 0, 1.f, stream);
    k_unpatch<<<2048, 256, 0, stream>>>(p.AO, outRecon);
}
